// Agent_65859028517180
// MI455X (gfx1250) — compile-verified, matrix operand order checked by reading, not run on hardware
//
#include <hip/hip_runtime.h>
#include <math.h>

// ---------------------------------------------------------------------------
// MI455X (gfx1250) fully-fused actor-critic forward.
//  * prologue kernel: pre-normalize (_wn) + bf16-convert all weights ONCE into
//    d_ws, pre-packed in the per-lane WMMA B-fragment layout (32B/lane).
//  * main kernel: one workgroup = 16 samples, 256 threads (8 wave32 waves).
//    All GEMMs -> v_wmma_f32_16x16x32_bf16; B-fragments are single vector
//    loads; per-block weight packs are DMA'd into LDS with the Tensor Data
//    Mover (tensor_load_to_lds + s_wait_tensorcnt), overlapping compute.
//  * x is read exactly once from HBM; all intermediates live in LDS.
// ---------------------------------------------------------------------------

typedef __bf16 v16bf __attribute__((ext_vector_type(16)));
typedef float  v8f   __attribute__((ext_vector_type(8)));
typedef unsigned v4u __attribute__((ext_vector_type(4)));
typedef int      v8i __attribute__((ext_vector_type(8)));
typedef int      v4i __attribute__((ext_vector_type(4)));

#if __has_builtin(__builtin_amdgcn_tensor_load_to_lds) && __has_builtin(__builtin_amdgcn_s_wait_tensorcnt)
#define USE_TDM 1
#endif

#define NIN 47
enum {
  IN_X = 0, IN_ACTION,
  P_AFC1_W, P_AFC1_B, P_ANORM1,
  P_ATOK_W, P_ATOK_B,
  P_EXPERT,
  TB1_QKV, TB1_O, TB1_GV, TB1_F, TB1_AA, TB1_AF,
  TB2_QKV, TB2_O, TB2_GV, TB2_F, TB2_AA, TB2_AF,
  P_AAGG_W, P_AAGG_B, P_AAGG_N,
  P_AOUT_W, P_AOUT_B,
  P_EMEAN_W, P_EMEAN_B,
  P_ELSD_W, P_ELSD_B,
  P_ROUTE_W, P_ROUTE_B,
  P_CFC1_W, P_CFC1_B, P_CNORM1,
  P_CTOK_W, P_CTOK_B,
  CTB_QKV, CTB_O, CTB_GV, CTB_F, CTB_AA, CTB_AF,
  P_CAGG_W, P_CAGG_B, P_CAGG_N,
  P_VOUT_W, P_VOUT_B
};

struct KArgs { const float* p[NIN]; const __bf16* wpk; float* out; int B; };

#define OBS_ 376
#define ACT_ 17

// ws pack layout (elements of __bf16; each tile = 512 elements = 1KB):
//  fc1 : tiles [path(2)][nt(4)][kc(12)]          -> tile ((path*4+nt)*12+kc)
//  tok : tiles [path(2)][nt(16)][kc(2)]          -> 96 + ((path*16+nt)*2+kc)
//  blk : 3 packs of 20 tiles (qkv 0-5, o 6-7, gv 8-15, f 16+n*2+kc)
#define WS_TOK_T   96
#define WS_BLK_T   160
#define TILE_EL    512
#define BLK_TILES  20
#define BLK_EL     (BLK_TILES * TILE_EL)   // 10240 elements = 20KB
#define WS_TOTAL_T (WS_BLK_T + 3 * BLK_TILES)

// ---------------- shared memory layout ----------------
struct Smem {
  float  T[192][33];          // token activations (actor: 12 tok/sample, critic: 8)
  alignas(32) __bf16 Wpack[BLK_TILES][TILE_EL];  // current block's fragment pack
  float  aat[32], aff[32];    // alphas
  union {
    __bf16 QKV[192][96];      // qkv output
    __bf16 GV[192][130];      // gate/value output
    float  TK[16][256];       // tokenize raw output
    float  P2[192][33];       // ffn_out raw output
  } u1;
  union {
    struct { __bf16 O[192][32]; float P[192][33]; } a;  // attn out + o_proj raw
    __bf16 Hf[192][66];       // silu(g)*v (ffn hidden), bf16
  } u2;
  float H1[2][16][64];        // fc1 hidden, both paths
  float pooled[16][32];
  float hh[16][64];
  float bm[16][ACT_];
  float sc[16][4];
  float pd[16][4];
};

// ---------------- WMMA fragment helpers (16x16x32 bf16 layout) -------------
// A (16x32, 16-bit): lane = half*16+idx ; row M=idx ; elem j -> K = kmap(j,half)
// B (32x16, 16-bit): lane = half*16+idx ; col N=idx ; elem j -> K = kmap(j,half)
// C/D (16x16 f32): lane col N=idx ; vgpr r -> row M = half*8 + r
__device__ __forceinline__ int kmap(int j, int half) {
  return (j < 8 ? j : j + 8) + half * 8;
}

__device__ __forceinline__ v8f wmma_bf16(v16bf a, v16bf b, v8f c) {
  return __builtin_amdgcn_wmma_f32_16x16x32_bf16(false, a, false, b,
                                                 (short)0, c, false, false);
}

// pre-packed fragment: lane's 16 bf16 are contiguous (32B aligned vector load)
__device__ __forceinline__ v16bf ldfrag(const __bf16* tile) {
  const int lane = threadIdx.x & 31;
  return *reinterpret_cast<const v16bf*>(tile + lane * 16);
}

// A fragment from f32 LDS (rows r0..r0+15, cols c0..c0+31), cvt to bf16
__device__ __forceinline__ v16bf afrag_f32(const float* p, int ld, int r0, int c0) {
  const int lane = threadIdx.x & 31, half = lane >> 4, idx = lane & 15;
  const float* row = p + (r0 + idx) * ld + c0;
  v16bf f;
#pragma unroll
  for (int j = 0; j < 16; ++j) f[j] = (__bf16)row[kmap(j, half)];
  return f;
}

// A fragment from bf16 LDS
__device__ __forceinline__ v16bf afrag_bf(const __bf16* p, int ld, int r0, int c0) {
  const int lane = threadIdx.x & 31, half = lane >> 4, idx = lane & 15;
  const __bf16* row = p + (r0 + idx) * ld + c0;
  v16bf f;
#pragma unroll
  for (int j = 0; j < 16; ++j) f[j] = row[kmap(j, half)];
  return f;
}

// A fragment straight from global f32 with column guard (x rows; K tail pad)
__device__ __forceinline__ v16bf frag_g(const float* p, int ld, int r0, int c0, int cmax) {
  const int lane = threadIdx.x & 31, half = lane >> 4, idx = lane & 15;
  const float* row = p + (size_t)(r0 + idx) * (size_t)ld;
  v16bf f;
#pragma unroll
  for (int j = 0; j < 16; ++j) {
    int c = c0 + kmap(j, half);
    f[j] = (c < cmax) ? (__bf16)row[c] : (__bf16)0.0f;
  }
  return f;
}

__device__ __forceinline__ void store_d_f32(float* p, int ld, int r0, int c0, v8f d) {
  const int lane = threadIdx.x & 31, half = lane >> 4, n = lane & 15;
#pragma unroll
  for (int r = 0; r < 8; ++r) p[(r0 + half * 8 + r) * ld + c0 + n] = d[r];
}

__device__ __forceinline__ void store_d_bf(__bf16* p, int ld, int r0, int c0, v8f d) {
  const int lane = threadIdx.x & 31, half = lane >> 4, n = lane & 15;
#pragma unroll
  for (int r = 0; r < 8; ++r) p[(r0 + half * 8 + r) * ld + c0 + n] = (__bf16)d[r];
}

// ---------------- block-weight pack staging (TDM async or fallback) --------
__device__ __forceinline__ void pack_issue(Smem& sm, const __bf16* src) {
#ifdef USE_TDM
  if ((threadIdx.x >> 5) == 0) {
    // Build Tensor-DMA descriptor (D#): 1-D tile of BLK_EL 2-byte elements.
    unsigned lds = (unsigned)(size_t)(&sm.Wpack[0][0]);
    unsigned long long ga = (unsigned long long)(size_t)src;
    v4u g0;
    g0[0] = 1u;                                            // count=1, user mode
    g0[1] = lds;                                           // lds_addr
    g0[2] = (unsigned)(ga & 0xffffffffull);                // global_addr[31:0]
    g0[3] = (unsigned)((ga >> 32) & 0x01ffffffull) | (2u << 30);  // [56:32] | type=2
    v8i g1;
    g1[0] = 1 << 16;                                       // data_size = 2B
    g1[1] = (BLK_EL & 0xffff) << 16;                       // tensor_dim0 lo16
    g1[2] = (BLK_EL >> 16) | (1 << 16);                    // tensor_dim0 hi16 | tensor_dim1=1
    g1[3] = (BLK_EL & 0xffff) << 16;                       // tile_dim0 (=10240)
    g1[4] = 1;                                             // tile_dim1=1, tile_dim2=0
    g1[5] = BLK_EL;                                        // tensor_dim0_stride
    g1[6] = 0; g1[7] = 0;
    v4i z4 = {0, 0, 0, 0};
    v8i z8 = {0, 0, 0, 0, 0, 0, 0, 0};
    __builtin_amdgcn_tensor_load_to_lds(g0, g1, z4, z4, z8, 0);
  }
#else
  for (int i = threadIdx.x; i < (int)(BLK_EL * 2 / 16); i += 256)
    ((uint4*)&sm.Wpack[0][0])[i] = ((const uint4*)src)[i];
#endif
}

__device__ __forceinline__ void pack_wait() {
#ifdef USE_TDM
  if ((threadIdx.x >> 5) == 0) __builtin_amdgcn_s_wait_tensorcnt(0);
#endif
  __syncthreads();
}

// x = l2norm(x + alpha*(l2norm(P) - x)), rows row-parallel
__device__ void row_update(Smem& sm, const float* P, int ldp, const float* alpha, int rows) {
  for (int row = threadIdx.x; row < rows; row += 256) {
    const float* pr = P + row * ldp;
    float ss = 0.f;
#pragma unroll
    for (int c = 0; c < 32; ++c) { float v = pr[c]; ss += v * v; }
    float hinv = 1.f / fmaxf(sqrtf(ss), 1e-6f);
    float s2 = 0.f;
#pragma unroll
    for (int c = 0; c < 32; ++c) {
      float xo = sm.T[row][c];
      float yv = xo + alpha[c] * (pr[c] * hinv - xo);
      s2 += yv * yv;
    }
    float oinv = 1.f / fmaxf(sqrtf(s2), 1e-6f);
#pragma unroll
    for (int c = 0; c < 32; ++c) {
      float xo = sm.T[row][c];
      float yv = xo + alpha[c] * (pr[c] * hinv - xo);
      sm.T[row][c] = yv * oinv;
    }
  }
}

// ---------------- one transformer block (TPS tokens per sample) ------------
// Weights already resident in sm.Wpack (pack_wait done by caller).
// nextPack (may be null): issued via TDM after the last weight read so the
// DMA overlaps the final row_update (and, for the critic pack, the whole
// actor tail).
template <int TPS>
__device__ void run_block(Smem& sm, const float* aa_g, const float* af_g,
                          const __bf16* nextPack) {
  const int tid = threadIdx.x, wid = tid >> 5;
  constexpr int ROWS = 16 * TPS, MT = ROWS / 16;

  if (tid < 32) { sm.aat[tid] = aa_g[tid]; sm.aff[tid] = af_g[tid]; }
  __syncthreads();

  // ---- qkv GEMM: [ROWS,32] x [32,96]  (input already unit-norm rows) ----
  for (int t = wid; t < MT * 6; t += 8) {
    int m = t / 6, n = t % 6;
    v16bf a = afrag_f32(&sm.T[0][0], 33, m * 16, 0);
    v16bf b = ldfrag(&sm.Wpack[n][0]);
    v8f d = {0.f, 0.f, 0.f, 0.f, 0.f, 0.f, 0.f, 0.f};
    d = wmma_bf16(a, b, d);
    store_d_bf(&sm.u1.QKV[0][0], 96, m * 16, n * 16, d);
  }
  __syncthreads();

  // ---- attention: 4 heads, HD=8, TPS x TPS, bf16 like the reference ----
  for (int it = tid; it < 16 * 4 * TPS; it += 256) {
    int s = it / (4 * TPS), r = it % (4 * TPS);
    int h = r / TPS, tq = r % TPS, rb = s * TPS;
    float q[8];
#pragma unroll
    for (int d0 = 0; d0 < 8; ++d0) q[d0] = (float)sm.u1.QKV[rb + tq][h * 8 + d0];
    float sco[TPS]; float mx = -1e30f;
#pragma unroll
    for (int tk = 0; tk < TPS; ++tk) {
      float acc = 0.f;
#pragma unroll
      for (int d0 = 0; d0 < 8; ++d0)
        acc += q[d0] * (float)sm.u1.QKV[rb + tk][32 + h * 8 + d0];
      sco[tk] = acc * 0.35355339059327373f;  // 1/sqrt(8)
      mx = fmaxf(mx, sco[tk]);
    }
    float ssum = 0.f;
#pragma unroll
    for (int tk = 0; tk < TPS; ++tk) { sco[tk] = __expf(sco[tk] - mx); ssum += sco[tk]; }
    float inv = 1.f / ssum;
    float o[8] = {0.f, 0.f, 0.f, 0.f, 0.f, 0.f, 0.f, 0.f};
#pragma unroll
    for (int tk = 0; tk < TPS; ++tk) {
      float pb = (float)(__bf16)(sco[tk] * inv);  // attn cast to bf16 (ref)
#pragma unroll
      for (int d0 = 0; d0 < 8; ++d0)
        o[d0] += pb * (float)sm.u1.QKV[rb + tk][64 + h * 8 + d0];
    }
#pragma unroll
    for (int d0 = 0; d0 < 8; ++d0)
      sm.u2.a.O[rb + tq][h * 8 + d0] = (__bf16)o[d0];
  }
  __syncthreads();

  // ---- o_proj GEMM: [ROWS,32] x [32,32] -> P (f32) ----
  for (int t = wid; t < MT * 2; t += 8) {
    int m = t >> 1, n = t & 1;
    v16bf a = afrag_bf(&sm.u2.a.O[0][0], 32, m * 16, 0);
    v16bf b = ldfrag(&sm.Wpack[6 + n][0]);
    v8f d = {0.f, 0.f, 0.f, 0.f, 0.f, 0.f, 0.f, 0.f};
    d = wmma_bf16(a, b, d);
    store_d_f32(&sm.u2.a.P[0][0], 33, m * 16, n * 16, d);
  }
  __syncthreads();
  row_update(sm, &sm.u2.a.P[0][0], 33, sm.aat, ROWS);
  __syncthreads();

  // ---- ffn gate/value GEMM: [ROWS,32] x [32,128] -> GV (bf16) ----
  for (int t = wid; t < MT * 8; t += 8) {
    int m = t >> 3, n = t & 7;
    v16bf a = afrag_f32(&sm.T[0][0], 33, m * 16, 0);
    v16bf b = ldfrag(&sm.Wpack[8 + n][0]);
    v8f d = {0.f, 0.f, 0.f, 0.f, 0.f, 0.f, 0.f, 0.f};
    d = wmma_bf16(a, b, d);
    store_d_bf(&sm.u1.GV[0][0], 130, m * 16, n * 16, d);
  }
  __syncthreads();

  // ---- h = silu(gate) * value -> Hf (bf16) ----
  for (int i = tid; i < ROWS * 64; i += 256) {
    int row = i >> 6, j = i & 63;
    float g = (float)sm.u1.GV[row][j];
    float v = (float)sm.u1.GV[row][64 + j];
    sm.u2.Hf[row][j] = (__bf16)((g / (1.f + __expf(-g))) * v);
  }
  __syncthreads();

  // ---- ffn_out GEMM: [ROWS,64] x [64,32] -> P2 (f32) ----
  for (int t = wid; t < MT * 2; t += 8) {
    int m = t >> 1, n = t & 1;
    v8f d = {0.f, 0.f, 0.f, 0.f, 0.f, 0.f, 0.f, 0.f};
#pragma unroll
    for (int kc = 0; kc < 2; ++kc) {
      v16bf a = afrag_bf(&sm.u2.Hf[0][0], 66, m * 16, kc * 32);
      v16bf b = ldfrag(&sm.Wpack[16 + n * 2 + kc][0]);
      d = wmma_bf16(a, b, d);
    }
    store_d_f32(&sm.u1.P2[0][0], 33, m * 16, n * 16, d);
  }
  __syncthreads();                // all weight reads done
  if (nextPack) pack_issue(sm, nextPack);   // overlap DMA with what follows
  row_update(sm, &sm.u1.P2[0][0], 33, sm.aff, ROWS);
  __syncthreads();
}

// ---------------- tokenize (fc1-hidden -> 8 unit-norm tokens) --------------
template <int TPS>
__device__ void tokenize(Smem& sm, const KArgs& A, int path) {
  const int tid = threadIdx.x, wid = tid >> 5;
  const __bf16* Wt = A.wpk + (size_t)WS_TOK_T * TILE_EL + (size_t)path * 32 * TILE_EL;
  const float* bt = A.p[path ? P_CTOK_B : P_ATOK_B];
  for (int t = wid; t < 16; t += 8) {      // 16 N-tiles of 16 cols
    v8f d = {0.f, 0.f, 0.f, 0.f, 0.f, 0.f, 0.f, 0.f};
#pragma unroll
    for (int kc = 0; kc < 2; ++kc) {
      v16bf a = afrag_f32(&sm.H1[path][0][0], 64, 0, kc * 32);
      v16bf b = ldfrag(Wt + (size_t)(t * 2 + kc) * TILE_EL);
      d = wmma_bf16(a, b, d);
    }
    store_d_f32(&sm.u1.TK[0][0], 256, 0, t * 16, d);
  }
  __syncthreads();
  if (tid < 128) {                         // bias + per-token l2norm
    int s = tid >> 3, tok = tid & 7;
    float ss = 0.f;
#pragma unroll
    for (int c = 0; c < 32; ++c) {
      float u = sm.u1.TK[s][tok * 32 + c] + bt[tok * 32 + c];
      ss += u * u;
    }
    float inv = 1.f / fmaxf(sqrtf(ss), 1e-6f);
#pragma unroll
    for (int c = 0; c < 32; ++c)
      sm.T[s * TPS + tok][c] = (sm.u1.TK[s][tok * 32 + c] + bt[tok * 32 + c]) * inv;
  }
}

// ---------------- prologue: build all weight packs in ws -------------------
__global__ void __launch_bounds__(32)
prep_weights(KArgs A, __bf16* ws) {
  const int lane = threadIdx.x & 31, half = lane >> 4, idx = lane & 15;
  const int bid = blockIdx.x;
  __bf16* dst = ws + (size_t)bid * TILE_EL + lane * 16;
  const float* src; int ld, row, k0, klen; bool norm = false;
  if (bid < WS_TOK_T) {                       // fc1 (not wn-normalized)
    int path = bid / 48, rem = bid % 48, nt = rem / 12, kc = rem % 12;
    src = A.p[path ? P_CFC1_W : P_AFC1_W]; ld = OBS_;
    row = nt * 16 + idx; k0 = kc * 32; klen = OBS_;
  } else if (bid < WS_BLK_T) {                // tokenize (not wn-normalized)
    int t = bid - WS_TOK_T, path = t / 32, rem = t % 32, nt = rem / 2, kc = rem & 1;
    src = A.p[path ? P_CTOK_W : P_ATOK_W]; ld = 64;
    row = nt * 16 + idx; k0 = kc * 32; klen = 64;
  } else {                                    // block weights (wn-normalized)
    int t = bid - WS_BLK_T, blk = t / BLK_TILES, ti = t % BLK_TILES;
    int base = (blk == 0) ? TB1_QKV : (blk == 1) ? TB2_QKV : CTB_QKV;
    norm = true;
    if (ti < 6)       { src = A.p[base + 0]; ld = 32; row = ti * 16 + idx;        k0 = 0;                  klen = 32; }
    else if (ti < 8)  { src = A.p[base + 1]; ld = 32; row = (ti - 6) * 16 + idx;  k0 = 0;                  klen = 32; }
    else if (ti < 16) { src = A.p[base + 2]; ld = 32; row = (ti - 8) * 16 + idx;  k0 = 0;                  klen = 32; }
    else              { src = A.p[base + 3]; ld = 64; row = ((ti - 16) >> 1) * 16 + idx;
                        k0 = ((ti - 16) & 1) * 32;                                klen = 64; }
  }
  const float* r = src + (size_t)row * (size_t)ld;
  float inv = 1.f;
  if (norm) {
    float ss = 0.f;
    for (int k = 0; k < klen; ++k) { float v = r[k]; ss += v * v; }
    inv = 1.f / fmaxf(sqrtf(ss), 1e-6f);    // _wn: w / clip(||w||, 1e-6)
  }
#pragma unroll
  for (int j = 0; j < 16; ++j) {
    int k = k0 + kmap(j, half);
    dst[j] = (k < klen) ? (__bf16)(r[k] * inv) : (__bf16)0.0f;
  }
}

// ---------------- main fused kernel ----------------
__global__ void __launch_bounds__(256)
fused_forward(KArgs A) {
  __shared__ Smem sm;
  const int tid = threadIdx.x, wid = tid >> 5;
  const int s0 = blockIdx.x * 16;
  const int B = A.B;
  const float* x   = A.p[IN_X];
  const float* act = A.p[IN_ACTION];
  const __bf16* blkpk = A.wpk + (size_t)WS_BLK_T * TILE_EL;

  // kick off tb1 weight-pack DMA; it overlaps fc1 + tokenize entirely
  pack_issue(sm, blkpk);

  // ======== fc1 for both actor & critic: [16,376] x [376,64] ========
  {
    int path = wid >> 2, nt = wid & 3;
    const __bf16* Wp = A.wpk + (size_t)((path * 4 + nt) * 12) * TILE_EL;
    v8f d = {0.f, 0.f, 0.f, 0.f, 0.f, 0.f, 0.f, 0.f};
    for (int kc = 0; kc < 12; ++kc) {        // 12*32 = 384 >= 376 (tail padded)
      v16bf a = frag_g(x, OBS_, s0, kc * 32, OBS_);
      v16bf b = ldfrag(Wp + (size_t)kc * TILE_EL);
      d = wmma_bf16(a, b, d);
    }
    store_d_f32(&sm.H1[path][0][0], 64, 0, nt * 16, d);
  }
  __syncthreads();
  // bias + rmsnorm + silu
  if (tid < 32) {
    int path = tid >> 4, s = tid & 15;
    const float* bias = A.p[path ? P_CFC1_B : P_AFC1_B];
    const float* nw   = A.p[path ? P_CNORM1 : P_ANORM1];
    float ss = 0.f;
#pragma unroll
    for (int o = 0; o < 64; ++o) {
      float u = sm.H1[path][s][o] + bias[o];
      ss += u * u;
    }
    float rms = sqrtf(ss * (1.f / 64.f) + 1e-6f);
#pragma unroll
    for (int o = 0; o < 64; ++o) {
      float u = (sm.H1[path][s][o] + bias[o]) / rms * nw[o];
      sm.H1[path][s][o] = u / (1.f + __expf(-u));
    }
  }
  __syncthreads();

  // ======== ACTOR path ========
  tokenize<12>(sm, A, 0);
  // expert tokens (l2norm, broadcast to every sample)
  if (tid >= 128 && tid < 192) {
    int t = tid - 128;
    int s = t >> 2, e = t & 3;
    const float* et = A.p[P_EXPERT] + e * 32;
    float ss = 0.f;
#pragma unroll
    for (int c = 0; c < 32; ++c) ss += et[c] * et[c];
    float inv = 1.f / fmaxf(sqrtf(ss), 1e-6f);
#pragma unroll
    for (int c = 0; c < 32; ++c) sm.T[s * 12 + 8 + e][c] = et[c] * inv;
  }
  pack_wait();   // tb1 pack ready (includes __syncthreads for token writes)

  run_block<12>(sm, A.p[TB1_AA], A.p[TB1_AF], blkpk + BLK_EL);       // tb1, prefetch tb2
  pack_wait();
  run_block<12>(sm, A.p[TB2_AA], A.p[TB2_AF], blkpk + 2 * BLK_EL);   // tb2, prefetch ctb

  // ---- actor tail (critic weight DMA in flight underneath) ----
  for (int t = tid; t < 16 * 32; t += 256) {   // pooled = mean of 8 policy tokens
    int s = t >> 5, c = t & 31;
    float acc = 0.f;
#pragma unroll
    for (int k = 0; k < 8; ++k) acc += sm.T[s * 12 + k][c];
    sm.pooled[s][c] = acc * 0.125f;
  }
  __syncthreads();
  {
    const float* Wg = A.p[P_AAGG_W]; const float* bg = A.p[P_AAGG_B];
    for (int t = tid; t < 16 * 64; t += 256) {
      int s = t >> 6, o = t & 63;
      float acc = bg[o];
#pragma unroll
      for (int c = 0; c < 32; ++c) acc += sm.pooled[s][c] * Wg[o * 32 + c];
      sm.hh[s][o] = acc;
    }
  }
  __syncthreads();
  if (tid < 16) {                               // rmsnorm + silu on hh
    const float* nw = A.p[P_AAGG_N];
    int s = tid; float ss = 0.f;
#pragma unroll
    for (int o = 0; o < 64; ++o) { float u = sm.hh[s][o]; ss += u * u; }
    float rms = sqrtf(ss * (1.f / 64.f) + 1e-6f);
#pragma unroll
    for (int o = 0; o < 64; ++o) {
      float u = sm.hh[s][o] / rms * nw[o];
      sm.hh[s][o] = u / (1.f + __expf(-u));
    }
  }
  __syncthreads();
  {
    const float* Wo_ = A.p[P_AOUT_W]; const float* bo = A.p[P_AOUT_B];
    for (int t = tid; t < 16 * ACT_; t += 256) {
      int s = t / ACT_, a = t % ACT_;
      float acc = bo[a];
#pragma unroll
      for (int c = 0; c < 64; ++c) acc += sm.hh[s][c] * Wo_[a * 64 + c];
      sm.bm[s][a] = acc;
    }
  }
  __syncthreads();
  // per (sample, expert): score + gaussian log-density sum over ACT dims
  if (tid < 64) {
    int s = tid >> 2, e = tid & 3;
    const float* eo = &sm.T[s * 12 + 8 + e][0];
    const float* rw = A.p[P_ROUTE_W];
    float score = A.p[P_ROUTE_B][0];
#pragma unroll
    for (int c = 0; c < 32; ++c) score += rw[c] * eo[c];
    sm.sc[s][e] = score;
    const float* mw = A.p[P_EMEAN_W]; const float* mb = A.p[P_EMEAN_B];
    const float* lw = A.p[P_ELSD_W];  const float* lb = A.p[P_ELSD_B];
    float sum = 0.f;
    for (int a = 0; a < ACT_; ++a) {
      float mo = mb[a], ls = lb[a];
#pragma unroll
      for (int c = 0; c < 32; ++c) {
        mo += mw[a * 32 + c] * eo[c];
        ls += lw[a * 32 + c] * eo[c];
      }
      ls = fminf(fmaxf(ls, -5.f), 2.f);
      float mean = sm.bm[s][a] + mo;
      float z = (act[(size_t)(s0 + s) * ACT_ + a] - mean) * __expf(-ls);
      sum += -0.5f * z * z - ls - 0.9189385332046727f;  // 0.5*log(2*pi)
    }
    sm.pd[s][e] = sum;
  }
  __syncthreads();
  if (tid < 16) {                               // log-softmax + logsumexp
    int s = tid;
    float m1 = sm.sc[s][0];
#pragma unroll
    for (int e = 1; e < 4; ++e) m1 = fmaxf(m1, sm.sc[s][e]);
    float z = 0.f;
#pragma unroll
    for (int e = 0; e < 4; ++e) z += __expf(sm.sc[s][e] - m1);
    float lz = m1 + logf(z);
    float le[4];
#pragma unroll
    for (int e = 0; e < 4; ++e) le[e] = (sm.sc[s][e] - lz) + sm.pd[s][e];
    float m2 = le[0];
#pragma unroll
    for (int e = 1; e < 4; ++e) m2 = fmaxf(m2, le[e]);
    float z2 = 0.f;
#pragma unroll
    for (int e = 0; e < 4; ++e) z2 += __expf(le[e] - m2);
    float lp = m2 + logf(z2);
    A.out[(size_t)B * 17 + (s0 + s)] = lp;      // log_prob
    A.out[(size_t)B * 18 + (s0 + s)] = -lp;     // entropy
  }
  // action pass-through
  for (int t = tid; t < 16 * ACT_; t += 256) {
    int s = t / ACT_, a = t % ACT_;
    size_t g = (size_t)(s0 + s) * ACT_ + a;
    A.out[g] = act[g];
  }
  __syncthreads();

  // ======== CRITIC path ========
  tokenize<8>(sm, A, 1);
  pack_wait();   // ctb pack ready (DMA overlapped the whole actor tail)
  run_block<8>(sm, A.p[CTB_AA], A.p[CTB_AF], nullptr);

  for (int t = tid; t < 16 * 32; t += 256) {    // cpool = mean of 8 tokens
    int s = t >> 5, c = t & 31;
    float acc = 0.f;
#pragma unroll
    for (int k = 0; k < 8; ++k) acc += sm.T[s * 8 + k][c];
    sm.pooled[s][c] = acc * 0.125f;
  }
  __syncthreads();
  {
    const float* Wg = A.p[P_CAGG_W]; const float* bg = A.p[P_CAGG_B];
    for (int t = tid; t < 16 * 64; t += 256) {
      int s = t >> 6, o = t & 63;
      float acc = bg[o];
#pragma unroll
      for (int c = 0; c < 32; ++c) acc += sm.pooled[s][c] * Wg[o * 32 + c];
      sm.hh[s][o] = acc;
    }
  }
  __syncthreads();
  if (tid < 16) {
    const float* nw = A.p[P_CAGG_N];
    int s = tid; float ss = 0.f;
#pragma unroll
    for (int o = 0; o < 64; ++o) { float u = sm.hh[s][o]; ss += u * u; }
    float rms = sqrtf(ss * (1.f / 64.f) + 1e-6f);
    float v = A.p[P_VOUT_B][0];
    const float* vw = A.p[P_VOUT_W];
#pragma unroll
    for (int o = 0; o < 64; ++o) {
      float u = sm.hh[s][o] / rms * nw[o];
      u = u / (1.f + __expf(-u));
      v += u * vw[o];
    }
    A.out[(size_t)B * 19 + (s0 + s)] = v;       // value
  }
}

// ---------------- launcher ----------------
extern "C" void kernel_launch(void* const* d_in, const int* in_sizes, int n_in,
                              void* d_out, int out_size, void* d_ws, size_t ws_size,
                              hipStream_t stream) {
  (void)ws_size; (void)out_size;
  KArgs A;
  for (int i = 0; i < NIN && i < n_in; ++i) A.p[i] = (const float*)d_in[i];
  A.wpk = (const __bf16*)d_ws;
  A.out = (float*)d_out;
  A.B = in_sizes[0] / OBS_;
  prep_weights<<<WS_TOTAL_T, 32, 0, stream>>>(A, (__bf16*)d_ws);
  fused_forward<<<A.B / 16, 256, 0, stream>>>(A);
}
